// LatentBasisResidualModel_64235530879189
// MI455X (gfx1250) — matrix operand order repaired and checked by reading, not compiled, not run on hardware
//
#include <hip/hip_runtime.h>
#include <math.h>

typedef __bf16 bf16;
typedef __attribute__((ext_vector_type(16))) __bf16 v16bf;
typedef __attribute__((ext_vector_type(8)))  float  v8f;

#define LN_EPS 1e-5f
#define L2_EPS 1e-12f
#define ATT_SCALE 0.03608439182435161f  /* 1/sqrt(768) */

// Problem constants
#define BB 32
#define NN 4096
#define DD 768
#define KK 64
#define HH 3072
#define CC 1000
#define BN (BB*NN)      /* 131072 */

#define WMMA_BF16(a,b,c) __builtin_amdgcn_wmma_f32_16x16x32_bf16(false,(a),false,(b),(short)0,(c),false,false)

// ---------------- reduction helpers (blockDim == 256, 8 waves of 32) -------------
__device__ __forceinline__ float block_sum(float v, volatile float* red) {
#pragma unroll
  for (int o = 16; o > 0; o >>= 1) v += __shfl_xor(v, o, 32);
  int wid = threadIdx.x >> 5, lane = threadIdx.x & 31;
  __syncthreads();
  if (lane == 0) red[wid] = v;
  __syncthreads();
  float r = 0.f;
#pragma unroll
  for (int i = 0; i < 8; ++i) r += red[i];
  return r;
}

__device__ __forceinline__ float block_max(float v, volatile float* red) {
#pragma unroll
  for (int o = 16; o > 0; o >>= 1) v = fmaxf(v, __shfl_xor(v, o, 32));
  int wid = threadIdx.x >> 5, lane = threadIdx.x & 31;
  __syncthreads();
  if (lane == 0) red[wid] = v;
  __syncthreads();
  float r = red[0];
#pragma unroll
  for (int i = 1; i < 8; ++i) r = fmaxf(r, red[i]);
  return r;
}

__device__ __forceinline__ float gelu_f(float x) {
  return 0.5f * x * (1.f + erff(x * 0.7071067811865476f));
}

// ---------------- WMMA fragment loaders (16x16x32 bf16; A and B both use contiguous k = 16*half + e) -------------
// A 16x32: lanes 0-15 row M=lane, K 0..15;
//          lanes 16-31 same rows, K 16..31.
__device__ __forceinline__ v16bf load_a_bf16(const bf16* A, int row, int k0, int lda) {
  int lane = threadIdx.x & 31, half = lane >> 4, r = lane & 15;
  const bf16* p = A + (size_t)(row + r) * lda + k0 + 16 * half;
  v16bf a;
#pragma unroll
  for (int e = 0; e < 8; ++e) a[e] = p[e];
#pragma unroll
  for (int e = 0; e < 8; ++e) a[8 + e] = p[8 + e];
  return a;
}

// B 32x16: lane n = lane&15; lanes 0-15 hold K 0..15, lanes 16-31 hold K 16..31.
__device__ __forceinline__ v16bf load_b_bf16(const bf16* Bw, int k0, int col, int ldb) {
  int lane = threadIdx.x & 31, half = lane >> 4, n = lane & 15;
  const bf16* p = Bw + (size_t)(k0 + 16 * half) * ldb + col + n;
  v16bf b;
#pragma unroll
  for (int e = 0; e < 16; ++e) b[e] = p[(size_t)e * ldb];
  return b;
}

__device__ __forceinline__ v16bf load_b_bf16_guard(const bf16* Bw, int k0, int col, int ldb, int ncols) {
  int lane = threadIdx.x & 31, half = lane >> 4, n = lane & 15;
  v16bf b;
  if (col + n < ncols) {
    const bf16* p = Bw + (size_t)(k0 + 16 * half) * ldb + col + n;
#pragma unroll
    for (int e = 0; e < 16; ++e) b[e] = p[(size_t)e * ldb];
  } else {
#pragma unroll
    for (int e = 0; e < 16; ++e) b[e] = (bf16)0.f;
  }
  return b;
}

// B fragment where B[k][n] = X[k][n] (X row-major f32, strided by ld).
__device__ __forceinline__ v16bf load_b_f32_strided(const float* X, int k0, int col, int ld) {
  int lane = threadIdx.x & 31, half = lane >> 4, n = lane & 15;
  const float* p = X + (size_t)(k0 + 16 * half) * ld + col + n;
  v16bf b;
#pragma unroll
  for (int e = 0; e < 16; ++e) b[e] = (bf16)p[(size_t)e * ld];
  return b;
}

// C/D 16x16 f32: VGPR j holds row M = j + 8*(lane>>4), column N = lane&15.
__device__ __forceinline__ void store_c_f32(float* C, const v8f& acc, int row, int col, int ldc) {
  int lane = threadIdx.x & 31, half = lane >> 4, n = lane & 15;
#pragma unroll
  for (int j = 0; j < 8; ++j)
    C[(size_t)(row + j + 8 * half) * ldc + col + n] = acc[j];
}

__device__ __forceinline__ void store_c_bias(float* C, const v8f& acc, int row, int col, int ldc, float bi) {
  int lane = threadIdx.x & 31, half = lane >> 4, n = lane & 15;
#pragma unroll
  for (int j = 0; j < 8; ++j)
    C[(size_t)(row + j + 8 * half) * ldc + col + n] = acc[j] + bi;
}

// ---------------- kernels ----------------------------------------------------------

// queries = l2norm(LN(bq, qn)); Qp = bf16(queries * tn_g); qc[k]=sum(Qp row), qc[64+k]=q.tn_b
// block 64: qc[128..130] = sum(tn_g^2), sum(tn_g*tn_b), sum(tn_b^2); zero accum.
__global__ __launch_bounds__(256) void k_prep(const float* bq, const float* qn_g, const float* qn_b,
                                              const float* tn_g, const float* tn_b,
                                              bf16* Qp, float* qc, float* bqn, float* accum) {
  __shared__ float red[8];
  int tid = threadIdx.x;
  if (blockIdx.x == 64) {
    float cg2 = 0, cgb = 0, cb2 = 0;
#pragma unroll
    for (int i = 0; i < 3; ++i) {
      int d = tid + i * 256;
      float g = tn_g[d], bb = tn_b[d];
      cg2 += g * g; cgb += g * bb; cb2 += bb * bb;
    }
    cg2 = block_sum(cg2, red); cgb = block_sum(cgb, red); cb2 = block_sum(cb2, red);
    if (tid == 0) { qc[128] = cg2; qc[129] = cgb; qc[130] = cb2; accum[0] = 0.f; }
    return;
  }
  int k = blockIdx.x;
  const float* x = bq + (size_t)k * DD;
  float xv[3], s1 = 0, s2 = 0;
#pragma unroll
  for (int i = 0; i < 3; ++i) { int d = tid + i * 256; xv[i] = x[d]; s1 += xv[i]; s2 += xv[i] * xv[i]; }
  s1 = block_sum(s1, red); s2 = block_sum(s2, red);
  float m = s1 * (1.f / DD);
  float var = s2 * (1.f / DD) - m * m;
  float rs = rsqrtf(var + LN_EPS);
  float rawinv = 1.f / fmaxf(sqrtf(s2), L2_EPS);
  float yv[3], ny = 0;
#pragma unroll
  for (int i = 0; i < 3; ++i) {
    int d = tid + i * 256;
    float y = (xv[i] - m) * rs * qn_g[d] + qn_b[d];
    yv[i] = y; ny += y * y;
  }
  ny = block_sum(ny, red);
  float inv = 1.f / fmaxf(sqrtf(ny), L2_EPS);
  float sq = 0, qb = 0;
#pragma unroll
  for (int i = 0; i < 3; ++i) {
    int d = tid + i * 256;
    float q = yv[i] * inv;
    float qg = q * tn_g[d];
    Qp[(size_t)k * DD + d] = (bf16)qg;
    sq += qg; qb += q * tn_b[d];
    bqn[(size_t)k * DD + d] = xv[i] * rawinv;
  }
  sq = block_sum(sq, red); qb = block_sum(qb, red);
  if (tid == 0) { qc[k] = sq; qc[64 + k] = qb; }
}

__global__ void k_f32_to_bf16(const float* src, bf16* dst, int n) {
  int i = blockIdx.x * blockDim.x + threadIdx.x;
  int stride = gridDim.x * blockDim.x;
  for (; i < n; i += stride) dst[i] = (bf16)src[i];
}

// Fused: S[64, BN] = Qp[64,768] x patch^T  AND  per-token LN/l2norm stats.
// Each wave owns 16 token columns; 4 accumulators cover all 64 query rows.
// Lane n and lane n+16 hold complementary K-halves of token (token_base+n), so the
// per-token reduction completes with one shfl_xor(16).
__global__ __launch_bounds__(256) void k_attn_gemm(const float* X, const bf16* Qp,
                                                   const float* tn_g, const float* tn_b,
                                                   const float* qc, float* S, float4* stats) {
  int w = threadIdx.x >> 5;
  int token_base = blockIdx.x * 128 + w * 16;
  int lane = threadIdx.x & 31, half = lane >> 4, n = lane & 15;
  v8f acc0 = {}, acc1 = {}, acc2 = {}, acc3 = {};
  float s1 = 0, s2 = 0, sxg2 = 0, sx2g2 = 0, sxgb = 0;
  const float* xrow = X + (size_t)(token_base + n) * DD;
  for (int kk = 0; kk < DD; kk += 32) {
    const float* p  = xrow + kk + 16 * half;
    const float* gp = tn_g + kk + 16 * half;
    const float* bp = tn_b + kk + 16 * half;
    v16bf b;
#pragma unroll
    for (int e = 0; e < 16; ++e) {
      float xv = p[e];
      b[e] = (bf16)xv;
      float g = gp[e], t = bp[e];
      s1 += xv; s2 += xv * xv;
      float xg = xv * g;
      sxg2 += xg * g; sx2g2 += xg * xg; sxgb += xg * t;
    }
    v16bf a0 = load_a_bf16(Qp, 0, kk, DD);
    v16bf a1 = load_a_bf16(Qp, 16, kk, DD);
    v16bf a2 = load_a_bf16(Qp, 32, kk, DD);
    v16bf a3 = load_a_bf16(Qp, 48, kk, DD);
    acc0 = WMMA_BF16(a0, b, acc0);
    acc1 = WMMA_BF16(a1, b, acc1);
    acc2 = WMMA_BF16(a2, b, acc2);
    acc3 = WMMA_BF16(a3, b, acc3);
  }
  store_c_f32(S, acc0, 0, token_base, BN);
  store_c_f32(S, acc1, 16, token_base, BN);
  store_c_f32(S, acc2, 32, token_base, BN);
  store_c_f32(S, acc3, 48, token_base, BN);
  // finish per-token stats
  s1 += __shfl_xor(s1, 16, 32);
  s2 += __shfl_xor(s2, 16, 32);
  sxg2 += __shfl_xor(sxg2, 16, 32);
  sx2g2 += __shfl_xor(sx2g2, 16, 32);
  sxgb += __shfl_xor(sxgb, 16, 32);
  if (half == 0) {
    float cg2 = qc[128], cgb = qc[129], cb2 = qc[130];
    float m = s1 * (1.f / DD);
    float var = s2 * (1.f / DD) - m * m;
    float rs = rsqrtf(var + LN_EPS);
    float n2 = rs * rs * (sx2g2 - 2.f * m * sxg2 + m * m * cg2) + 2.f * rs * (sxgb - m * cgb) + cb2;
    float invn = 1.f / fmaxf(sqrtf(fmaxf(n2, 0.f)), L2_EPS);
    stats[token_base + n] = make_float4(rs * invn, m * rs * invn, invn, 0.f);
  }
}

// logit fixup + softmax over N per (b,k); writes bf16 attention weights
__global__ __launch_bounds__(256) void k_softmax(const float* S, const float4* stats, const float* qc,
                                                 bf16* W) {
  __shared__ float red[8];
  int bk = blockIdx.x;
  int b = bk >> 6, k = bk & 63;
  float sqk = qc[k], qbk = qc[64 + k];
  const float* srow = S + (size_t)k * BN + (size_t)b * NN;
  const float4* st = stats + (size_t)b * NN;
  int tid = threadIdx.x;
  float lg[16], mx = -INFINITY;
#pragma unroll
  for (int i = 0; i < 16; ++i) {
    int n = tid + i * 256;
    float4 s = st[n];
    float l = ATT_SCALE * (s.x * srow[n] - s.y * sqk + s.z * qbk);
    lg[i] = l; mx = fmaxf(mx, l);
  }
  mx = block_max(mx, red);
  float sum = 0;
#pragma unroll
  for (int i = 0; i < 16; ++i) { lg[i] = expf(lg[i] - mx); sum += lg[i]; }
  sum = block_sum(sum, red);
  float inv = 1.f / sum;
  bf16* wr = W + (size_t)bk * NN;
#pragma unroll
  for (int i = 0; i < 16; ++i) { int n = tid + i * 256; wr[n] = (bf16)(lg[i] * inv); }
}

// basis[b,64,768] = W[b,64,4096] x patch[b,4096,768]; one B strip per wave, 4 M-tiles.
__global__ __launch_bounds__(256) void k_basis_gemm(const bf16* W, const float* X, float* Bs) {
  int b = blockIdx.y;
  int w = threadIdx.x >> 5;
  int col = blockIdx.x * 128 + w * 16;
  const bf16* A = W + (size_t)b * KK * NN;
  const float* Xb = X + (size_t)b * NN * DD;
  v8f acc0 = {}, acc1 = {}, acc2 = {}, acc3 = {};
  for (int kk = 0; kk < NN; kk += 32) {
    v16bf bb = load_b_f32_strided(Xb, kk, col, DD);
    v16bf a0 = load_a_bf16(A, 0, kk, NN);
    v16bf a1 = load_a_bf16(A, 16, kk, NN);
    v16bf a2 = load_a_bf16(A, 32, kk, NN);
    v16bf a3 = load_a_bf16(A, 48, kk, NN);
    acc0 = WMMA_BF16(a0, bb, acc0);
    acc1 = WMMA_BF16(a1, bb, acc1);
    acc2 = WMMA_BF16(a2, bb, acc2);
    acc3 = WMMA_BF16(a3, bb, acc3);
  }
  float* out = Bs + (size_t)b * KK * DD;
  store_c_f32(out, acc0, 0, col, DD);
  store_c_f32(out, acc1, 16, col, DD);
  store_c_f32(out, acc2, 32, col, DD);
  store_c_f32(out, acc3, 48, col, DD);
}

// row LN of basis -> bf16 R
__global__ __launch_bounds__(256) void k_rowln_bf16(const float* X, const float* g, const float* bb,
                                                    bf16* Y) {
  __shared__ float red[8];
  int row = blockIdx.x, tid = threadIdx.x;
  const float* x = X + (size_t)row * DD;
  float xv[3], s1 = 0, s2 = 0;
#pragma unroll
  for (int i = 0; i < 3; ++i) { int d = tid + i * 256; xv[i] = x[d]; s1 += xv[i]; s2 += xv[i] * xv[i]; }
  s1 = block_sum(s1, red); s2 = block_sum(s2, red);
  float m = s1 * (1.f / DD), var = s2 * (1.f / DD) - m * m, rs = rsqrtf(var + LN_EPS);
#pragma unroll
  for (int i = 0; i < 3; ++i) {
    int d = tid + i * 256;
    Y[(size_t)row * DD + d] = (bf16)((xv[i] - m) * rs * g[d] + bb[d]);
  }
}

// H1[2048,3072] = R[2048,768] x rw1[768,3072] + rb1; 4 M-tiles per wave.
__global__ __launch_bounds__(256) void k_routing_gemm(const bf16* A, const bf16* Bw, const float* bias,
                                                      float* H1) {
  int w = threadIdx.x >> 5;
  int rowblk = blockIdx.y * 64;
  int col = blockIdx.x * 128 + w * 16;
  v8f acc0 = {}, acc1 = {}, acc2 = {}, acc3 = {};
  for (int kk = 0; kk < DD; kk += 32) {
    v16bf b = load_b_bf16(Bw, kk, col, HH);
    v16bf a0 = load_a_bf16(A, rowblk + 0, kk, DD);
    v16bf a1 = load_a_bf16(A, rowblk + 16, kk, DD);
    v16bf a2 = load_a_bf16(A, rowblk + 32, kk, DD);
    v16bf a3 = load_a_bf16(A, rowblk + 48, kk, DD);
    acc0 = WMMA_BF16(a0, b, acc0);
    acc1 = WMMA_BF16(a1, b, acc1);
    acc2 = WMMA_BF16(a2, b, acc2);
    acc3 = WMMA_BF16(a3, b, acc3);
  }
  int lane = threadIdx.x & 31, n = lane & 15;
  float bi = bias[col + n];
  store_c_bias(H1, acc0, rowblk + 0, col, HH, bi);
  store_c_bias(H1, acc1, rowblk + 16, col, HH, bi);
  store_c_bias(H1, acc2, rowblk + 32, col, HH, bi);
  store_c_bias(H1, acc3, rowblk + 48, col, HH, bi);
}

// score[row] = softplus(sum_h gelu(H1[row,h]) * rw2[h] + rb2); sparse-loss accumulation
__global__ __launch_bounds__(256) void k_routing_reduce(const float* H1, const float* rw2, const float* rb2,
                                                        float* scores, float* accum) {
  __shared__ float red[8];
  int row = blockIdx.x, tid = threadIdx.x;
  const float* h = H1 + (size_t)row * HH;
  float acc = 0.f;
#pragma unroll
  for (int i = 0; i < 12; ++i) { int d = tid + i * 256; acc += gelu_f(h[d]) * rw2[d]; }
  acc = block_sum(acc, red);
  if (tid == 0) {
    float z = acc + rb2[0];
    float sp = fmaxf(z, 0.f) + log1pf(expf(-fabsf(z)));
    scores[row] = sp;
    atomicAdd(accum, sp * (1.f / (BB * KK)));
  }
}

// top-8 routing + basis summary LN + global LN -> fused bf16 [B,1536]; routing weights -> d_out
__global__ __launch_bounds__(256) void k_route(const float* scores, const float* basis, const float* gf,
                                               const float* bn_g, const float* bn_b,
                                               const float* gn_g, const float* gn_b,
                                               float* out, bf16* fused) {
  __shared__ float red[8];
  __shared__ float sc[64];
  __shared__ float wsh[64];
  int b = blockIdx.x, tid = threadIdx.x;
  if (tid < 64) { sc[tid] = scores[b * 64 + tid]; wsh[tid] = 0.f; }
  __syncthreads();
  if (tid == 0) {
    bool used[64];
    for (int i = 0; i < 64; ++i) used[i] = false;
    float vals[8]; int idxs[8];
    float denom = 0.f;
    for (int t = 0; t < 8; ++t) {
      int bi = 0; float bv = -INFINITY;
      for (int i = 0; i < 64; ++i)
        if (!used[i] && sc[i] > bv) { bv = sc[i]; bi = i; }
      used[bi] = true; vals[t] = bv; idxs[t] = bi; denom += bv;
    }
    denom = fmaxf(denom, 1e-8f);
    for (int t = 0; t < 8; ++t) wsh[idxs[t]] = vals[t] / denom;
  }
  __syncthreads();
  if (tid < 64) out[2 * BB * CC + b * 64 + tid] = wsh[tid];
  // basis summary
  float sv[3];
#pragma unroll
  for (int i = 0; i < 3; ++i) {
    int d = tid + i * 256;
    float s = 0.f;
    for (int k2 = 0; k2 < 64; ++k2) {
      float wk = wsh[k2];
      if (wk != 0.f) s += wk * basis[((size_t)b * 64 + k2) * DD + d];
    }
    sv[i] = s;
  }
  float s1 = 0, s2 = 0;
#pragma unroll
  for (int i = 0; i < 3; ++i) { s1 += sv[i]; s2 += sv[i] * sv[i]; }
  s1 = block_sum(s1, red); s2 = block_sum(s2, red);
  float m = s1 * (1.f / DD), var = s2 * (1.f / DD) - m * m, rs = rsqrtf(var + LN_EPS);
#pragma unroll
  for (int i = 0; i < 3; ++i) {
    int d = tid + i * 256;
    fused[(size_t)b * 1536 + DD + d] = (bf16)((sv[i] - m) * rs * bn_g[d] + bn_b[d]);
  }
  // global summary
  float gv[3], g1 = 0, g2 = 0;
#pragma unroll
  for (int i = 0; i < 3; ++i) { int d = tid + i * 256; gv[i] = gf[(size_t)b * DD + d]; g1 += gv[i]; g2 += gv[i] * gv[i]; }
  g1 = block_sum(g1, red); g2 = block_sum(g2, red);
  float mg = g1 * (1.f / DD), vg = g2 * (1.f / DD) - mg * mg, rg = rsqrtf(vg + LN_EPS);
#pragma unroll
  for (int i = 0; i < 3; ++i) {
    int d = tid + i * 256;
    fused[(size_t)b * 1536 + d] = (bf16)((gv[i] - mg) * rg * gn_g[d] + gn_b[d]);
  }
}

// H2[32,3072] = gelu(fused[32,1536] x hw1[1536,3072] + hb1); 2 M-tiles per wave.
__global__ __launch_bounds__(256) void k_head_gemm(const bf16* A, const bf16* Bw, const float* bias,
                                                   bf16* H2) {
  int w = threadIdx.x >> 5;
  int col = blockIdx.x * 128 + w * 16;
  v8f acc0 = {}, acc1 = {};
  for (int kk = 0; kk < 1536; kk += 32) {
    v16bf b = load_b_bf16(Bw, kk, col, HH);
    v16bf a0 = load_a_bf16(A, 0, kk, 1536);
    v16bf a1 = load_a_bf16(A, 16, kk, 1536);
    acc0 = WMMA_BF16(a0, b, acc0);
    acc1 = WMMA_BF16(a1, b, acc1);
  }
  int lane = threadIdx.x & 31, half = lane >> 4, n = lane & 15;
  int nc = col + n;
  float bi = bias[nc];
#pragma unroll
  for (int j = 0; j < 8; ++j) {
    int m0 = j + 8 * half;
    H2[(size_t)m0 * HH + nc] = (bf16)gelu_f(acc0[j] + bi);
    H2[(size_t)(m0 + 16) * HH + nc] = (bf16)gelu_f(acc1[j] + bi);
  }
}

// residual[32,1000] = H2 x hw2 + hb2 ; final = base + residual; 2 M-tiles per wave.
__global__ __launch_bounds__(256) void k_out_gemm(const bf16* A, const bf16* Bw, const float* hb2,
                                                  const float* base, float* out) {
  int w = threadIdx.x >> 5;
  int col = blockIdx.x * 128 + w * 16;
  v8f acc0 = {}, acc1 = {};
  for (int kk = 0; kk < HH; kk += 32) {
    v16bf b = load_b_bf16_guard(Bw, kk, col, CC, CC);
    v16bf a0 = load_a_bf16(A, 0, kk, HH);
    v16bf a1 = load_a_bf16(A, 16, kk, HH);
    acc0 = WMMA_BF16(a0, b, acc0);
    acc1 = WMMA_BF16(a1, b, acc1);
  }
  int lane = threadIdx.x & 31, half = lane >> 4, n = lane & 15;
  int nc = col + n;
  if (nc < CC) {
    float bi = hb2[nc];
#pragma unroll
    for (int j = 0; j < 8; ++j) {
      int m0 = j + 8 * half;
      float r0 = acc0[j] + bi;
      float r1 = acc1[j] + bi;
      out[BB * CC + m0 * CC + nc] = r0;
      out[m0 * CC + nc] = base[m0 * CC + nc] + r0;
      out[BB * CC + (m0 + 16) * CC + nc] = r1;
      out[(m0 + 16) * CC + nc] = base[(m0 + 16) * CC + nc] + r1;
    }
  }
}

// diversity loss over bqn gram + emit sparse loss
__global__ __launch_bounds__(256) void k_div(const float* bqn, const float* accum, float* out) {
  __shared__ float red[8];
  int tid = threadIdx.x;
  float loc = 0.f;
  for (int p = tid; p < 4096; p += 256) {
    int i = p >> 6, j = p & 63;
    const float* a = bqn + (size_t)i * DD;
    const float* c = bqn + (size_t)j * DD;
    float dot = 0.f;
    for (int d = 0; d < DD; ++d) dot += a[d] * c[d];
    float diff = dot - (i == j ? 1.f : 0.f);
    loc += diff * diff;
  }
  loc = block_sum(loc, red);
  if (tid == 0) {
    out[2 * BB * CC + BB * KK + 1] = loc * (1.f / 4096.f);  // diversity_loss
    out[2 * BB * CC + BB * KK]     = accum[0];              // sparse_loss
  }
}

// ---------------- workspace layout --------------------------------------------------
static constexpr size_t alignup(size_t x) { return (x + 255) & ~(size_t)255; }
static constexpr size_t OFF_STATS = 0;
static constexpr size_t OFF_S     = alignup(OFF_STATS + (size_t)BN * 4 * sizeof(float));
static constexpr size_t OFF_WBF   = alignup(OFF_S     + (size_t)KK * BN * sizeof(float));
static constexpr size_t OFF_BASIS = alignup(OFF_WBF   + (size_t)BB * KK * NN * sizeof(bf16));
static constexpr size_t OFF_RBF   = alignup(OFF_BASIS + (size_t)BB * KK * DD * sizeof(float));
static constexpr size_t OFF_H1    = alignup(OFF_RBF   + (size_t)BB * KK * DD * sizeof(bf16));
static constexpr size_t OFF_QP    = alignup(OFF_H1    + (size_t)BB * KK * HH * sizeof(float));
static constexpr size_t OFF_QC    = alignup(OFF_QP    + (size_t)KK * DD * sizeof(bf16));
static constexpr size_t OFF_RW1B  = alignup(OFF_QC    + 256 * sizeof(float));
static constexpr size_t OFF_HW1B  = alignup(OFF_RW1B  + (size_t)DD * HH * sizeof(bf16));
static constexpr size_t OFF_HW2B  = alignup(OFF_HW1B  + (size_t)2 * DD * HH * sizeof(bf16));
static constexpr size_t OFF_SCOR  = alignup(OFF_HW2B  + (size_t)HH * CC * sizeof(bf16));
static constexpr size_t OFF_FUSE  = alignup(OFF_SCOR  + (size_t)BB * KK * sizeof(float));
static constexpr size_t OFF_H2B   = alignup(OFF_FUSE  + (size_t)BB * 1536 * sizeof(bf16));
static constexpr size_t OFF_BQN   = alignup(OFF_H2B   + (size_t)BB * HH * sizeof(bf16));
static constexpr size_t OFF_ACC   = alignup(OFF_BQN   + (size_t)KK * DD * sizeof(float));

extern "C" void kernel_launch(void* const* d_in, const int* in_sizes, int n_in,
                              void* d_out, int out_size, void* d_ws, size_t ws_size,
                              hipStream_t stream) {
  const float* patch = (const float*)d_in[0];
  const float* gf    = (const float*)d_in[1];
  const float* base  = (const float*)d_in[2];
  const float* bq    = (const float*)d_in[3];
  const float* qn_g  = (const float*)d_in[4];
  const float* qn_b  = (const float*)d_in[5];
  const float* tn_g  = (const float*)d_in[6];
  const float* tn_b  = (const float*)d_in[7];
  const float* bn_g  = (const float*)d_in[8];
  const float* bn_b  = (const float*)d_in[9];
  const float* gn_g  = (const float*)d_in[10];
  const float* gn_b  = (const float*)d_in[11];
  const float* rln_g = (const float*)d_in[12];
  const float* rln_b = (const float*)d_in[13];
  const float* rw1   = (const float*)d_in[14];
  const float* rb1   = (const float*)d_in[15];
  const float* rw2   = (const float*)d_in[16];
  const float* rb2   = (const float*)d_in[17];
  const float* hw1   = (const float*)d_in[18];
  const float* hb1   = (const float*)d_in[19];
  const float* hw2   = (const float*)d_in[20];
  const float* hb2   = (const float*)d_in[21];
  float* out = (float*)d_out;
  char* ws = (char*)d_ws;

  float4* stats = (float4*)(ws + OFF_STATS);
  float* S      = (float*)(ws + OFF_S);
  bf16*  Wbf    = (bf16*)(ws + OFF_WBF);
  float* basis  = (float*)(ws + OFF_BASIS);
  bf16*  Rbf    = (bf16*)(ws + OFF_RBF);
  float* H1     = (float*)(ws + OFF_H1);
  bf16*  Qp     = (bf16*)(ws + OFF_QP);
  float* qc     = (float*)(ws + OFF_QC);
  bf16*  rw1bf  = (bf16*)(ws + OFF_RW1B);
  bf16*  hw1bf  = (bf16*)(ws + OFF_HW1B);
  bf16*  hw2bf  = (bf16*)(ws + OFF_HW2B);
  float* scores = (float*)(ws + OFF_SCOR);
  bf16*  fused  = (bf16*)(ws + OFF_FUSE);
  bf16*  H2bf   = (bf16*)(ws + OFF_H2B);
  float* bqn    = (float*)(ws + OFF_BQN);
  float* accum  = (float*)(ws + OFF_ACC);

  k_prep<<<65, 256, 0, stream>>>(bq, qn_g, qn_b, tn_g, tn_b, Qp, qc, bqn, accum);
  k_f32_to_bf16<<<2048, 256, 0, stream>>>(rw1, rw1bf, DD * HH);
  k_f32_to_bf16<<<2048, 256, 0, stream>>>(hw1, hw1bf, 2 * DD * HH);
  k_f32_to_bf16<<<2048, 256, 0, stream>>>(hw2, hw2bf, HH * CC);
  k_attn_gemm<<<BN / 128, 256, 0, stream>>>(patch, Qp, tn_g, tn_b, qc, S, stats);
  k_softmax<<<BB * KK, 256, 0, stream>>>(S, stats, qc, Wbf);
  { dim3 g(DD / 128, BB); k_basis_gemm<<<g, 256, 0, stream>>>(Wbf, patch, basis); }
  k_rowln_bf16<<<BB * KK, 256, 0, stream>>>(basis, rln_g, rln_b, Rbf);
  { dim3 g(HH / 128, BB * KK / 64); k_routing_gemm<<<g, 256, 0, stream>>>(Rbf, rw1bf, rb1, H1); }
  k_routing_reduce<<<BB * KK, 256, 0, stream>>>(H1, rw2, rb2, scores, accum);
  k_route<<<BB, 256, 0, stream>>>(scores, basis, gf, bn_g, bn_b, gn_g, gn_b, out, fused);
  k_head_gemm<<<HH / 128, 256, 0, stream>>>(fused, hw1bf, hb1, H2bf);
  k_out_gemm<<<8, 256, 0, stream>>>(H2bf, hw2bf, hb2, base, out);
  k_div<<<1, 256, 0, stream>>>(bqn, accum, out);
}
